// ReplicatedLinear_56195352101389
// MI455X (gfx1250) — matrix operand order repaired and checked by reading, not compiled, not run on hardware
//
#include <hip/hip_runtime.h>
#include <hip/hip_bf16.h>

// ---- types for the CDNA5 WMMA builtins ----
typedef __bf16 bf16_t;
typedef bf16_t v16bf __attribute__((ext_vector_type(16)));
typedef float  v8f   __attribute__((ext_vector_type(8)));
typedef unsigned int v4u __attribute__((ext_vector_type(4)));

struct B32x2 { v4u x, y; };            // 32 B == 16 bf16 == one WMMA fragment
struct H8    { unsigned short h[8]; }; // 16 B of packed bf16

__device__ __forceinline__ unsigned short f2bf_u16(float f) {
  unsigned u = __builtin_bit_cast(unsigned, f);
  unsigned r = u + 0x7FFFu + ((u >> 16) & 1u);   // round-to-nearest-even
  return (unsigned short)(r >> 16);
}
__device__ __forceinline__ bf16_t f2bf(float f) {
  return __builtin_bit_cast(bf16_t, f2bf_u16(f));
}

// Problem shape (fixed by the reference harness)
constexpr int M_TOK = 4096;
constexpr int K_DIM = 4096;
constexpr int N_DIM = 11008;
constexpr int GROUP = 128;

// =====================================================================
// Pass 1: x f32 -> bf16 (memory bound one-shot; VALU cost irrelevant)
// =====================================================================
__global__ __launch_bounds__(256)
void convert_x_bf16(const float* __restrict__ x, bf16_t* __restrict__ xb) {
  int i = (blockIdx.x * 256 + threadIdx.x) * 8;
  const float4* p = reinterpret_cast<const float4*>(x + i);
  float4 a = p[0], b = p[1];
  H8 h;
  h.h[0] = f2bf_u16(a.x); h.h[1] = f2bf_u16(a.y);
  h.h[2] = f2bf_u16(a.z); h.h[3] = f2bf_u16(a.w);
  h.h[4] = f2bf_u16(b.x); h.h[5] = f2bf_u16(b.y);
  h.h[6] = f2bf_u16(b.z); h.h[7] = f2bf_u16(b.w);
  *reinterpret_cast<v4u*>(xb + i) = __builtin_bit_cast(v4u, h);
}

// =====================================================================
// Pass 2: AWQ int4 -> bf16, W stored TRANSPOSED as [N][K] (one-shot).
// Transpose goes through LDS so global reads AND writes are coalesced.
// =====================================================================
__global__ __launch_bounds__(256)
void dequant_wt_bf16(const int*   __restrict__ qweight,
                     const float* __restrict__ scales,
                     const int*   __restrict__ qzeros,
                     bf16_t*      __restrict__ wt) {
  __shared__ __align__(16) bf16_t T[64][40];   // [n][k], padded row = 80 B
  const int tid = threadIdx.x;
  const int k0  = blockIdx.x * 32;             // K patch
  const int n0  = blockIdx.y * 64;             // N patch
  const int shifts[8] = {0, 16, 4, 20, 8, 24, 12, 28};   // AWQ nibble order

  const int k   = tid >> 3;                    // 0..31
  const int cg  = tid & 7;                     // 0..7
  const int grp = k0 / GROUP;                  // 32 | 128 -> single group
  unsigned q = (unsigned)qweight[(size_t)(k0 + k) * (N_DIM / 8) + n0 / 8 + cg];
  unsigned z = (unsigned)qzeros[(size_t)grp * (N_DIM / 8) + n0 / 8 + cg];
  const float* s = scales + (size_t)grp * N_DIM + n0 + cg * 8;
#pragma unroll
  for (int j = 0; j < 8; ++j) {
    float wv = (float)((q >> shifts[j]) & 15u);
    float zv = (float)((z >> shifts[j]) & 15u);
    T[cg * 8 + j][k] = f2bf((wv - zv) * s[j]);
  }
  __syncthreads();

  const int row = tid >> 2;                    // 0..63
  const int ch  = tid & 3;                     // 0..3
  v4u val = *reinterpret_cast<const v4u*>(&T[row][ch * 8]);
  *reinterpret_cast<v4u*>(wt + (size_t)(n0 + row) * K_DIM + k0 + ch * 8) = val;
}

// =====================================================================
// Pass 3: pure bf16 WMMA GEMM, async-to-LDS staging, double buffered.
// Block tile 128(M) x 256(N), wave tile 64x64 (4x4 WMMA tiles) so each
// fragment load feeds 4 WMMAs: 16 ds_load_b128 per 16 WMMAs per K-step.
// =====================================================================
constexpr int BM = 128, BN = 256, BK = 32;
constexpr int LDB = BK + 8;                       // padded row: 40 bf16 = 80 B
constexpr int A_TILE_BYTES = 128 * LDB * 2;       // 10240 B
constexpr int B_TILE_BYTES = 256 * LDB * 2;       // 20480 B
constexpr int NSTAGE = K_DIM / BK;                // 128

// Issue ITERS async b128 loads per thread (ROWS = ITERS*64 rows of 64 B).
// ASYNCcnt += ITERS per wave per call.
template <int ITERS>
__device__ __forceinline__ void async_stage_tile(unsigned lds_base,
                                                 const bf16_t* g0,
                                                 int tid) {
#pragma unroll
  for (int it = 0; it < ITERS; ++it) {
    int u   = tid + it * 256;
    int row = u >> 2;                         // 64-B row
    int seg = u & 3;                          // 16-B segment within row
    unsigned loff = lds_base + row * (LDB * 2) + seg * 16;
    const char* ga = (const char*)g0 + (size_t)row * (K_DIM * 2) + seg * 16;
    asm volatile("global_load_async_to_lds_b128 %0, %1, off"
                 :: "v"(loff), "v"(ga) : "memory");
  }
}

__global__ __launch_bounds__(256)
void gemm_bf16_async(const bf16_t* __restrict__ xb,   // [M][K] bf16
                     const bf16_t* __restrict__ wt,   // [N][K] bf16
                     float*        __restrict__ out) {
  __shared__ __align__(16) bf16_t As[2][128][LDB];    // 20 KB
  __shared__ __align__(16) bf16_t Bs[2][256][LDB];    // 40 KB

  const int tid  = threadIdx.x;
  const int lane = tid & 31;
  const int wave = tid >> 5;
  const int wm   = wave >> 2;    // 0..1 -> wave M offset = wm*64
  const int wn   = wave & 3;     // 0..3 -> wave N offset = wn*64
  const int n0   = blockIdx.x * BN;
  const int m0   = blockIdx.y * BM;

  const unsigned abase = (unsigned)(size_t)&As[0][0][0];
  const unsigned bbase = (unsigned)(size_t)&Bs[0][0][0];

  v8f acc[4][4];
#pragma unroll
  for (int i = 0; i < 4; ++i)
#pragma unroll
    for (int j = 0; j < 4; ++j)
#pragma unroll
      for (int v = 0; v < 8; ++v) acc[i][j][v] = 0.0f;

  const bf16_t* ga = xb + (size_t)m0 * K_DIM;   // A tile base (k advances)
  const bf16_t* gb = wt + (size_t)n0 * K_DIM;   // B tile base (k advances)

  // prologue: stage buffer 0 (2 + 4 = 6 async issues per thread)
  async_stage_tile<2>(abase, ga, tid);
  async_stage_tile<4>(bbase, gb, tid);

  const int r   = lane & 15;
  const int kbA = (lane < 16) ? 0 : 16;   // A: K 0..15 | 16..31
  const int kbB = (lane < 16) ? 0 : 16;   // B: K 0..15 | 16..31

  for (int t = 0; t < NSTAGE; ++t) {
    const int buf = t & 1;
    if (t + 1 < NSTAGE) {
      const int nb = (t + 1) & 1;
      async_stage_tile<2>(abase + nb * A_TILE_BYTES, ga + (size_t)(t + 1) * BK, tid);
      async_stage_tile<4>(bbase + nb * B_TILE_BYTES, gb + (size_t)(t + 1) * BK, tid);
      // 6 in flight from stage t+1; stage t's 6 are complete (in-order)
      asm volatile("s_wait_asynccnt 0x6" ::: "memory");
    } else {
      asm volatile("s_wait_asynccnt 0x0" ::: "memory");
    }
    __syncthreads();

    // 4 B fragments, reused by all 4 A tiles
    v16bf bfrag[4];
#pragma unroll
    for (int j = 0; j < 4; ++j) {
      const bf16_t* bp = &Bs[buf][wn * 64 + j * 16 + r][kbB];
      B32x2 tt;
      tt.x = *reinterpret_cast<const v4u*>(bp);
      tt.y = *reinterpret_cast<const v4u*>(bp + 8);
      bfrag[j] = __builtin_bit_cast(v16bf, tt);
    }

#pragma unroll
    for (int i = 0; i < 4; ++i) {
      const bf16_t* ap = &As[buf][wm * 64 + i * 16 + r][0];
      B32x2 tt;
      tt.x = *reinterpret_cast<const v4u*>(ap + kbA);
      tt.y = *reinterpret_cast<const v4u*>(ap + kbA + 8);
      v16bf afrag = __builtin_bit_cast(v16bf, tt);
#pragma unroll
      for (int j = 0; j < 4; ++j)
        acc[i][j] = __builtin_amdgcn_wmma_f32_16x16x32_bf16(
            false, afrag, false, bfrag[j], (short)0, acc[i][j], false, false);
    }

    __syncthreads();   // safe to overwrite this buffer next time around
  }

  // writeback per 16x16 f32 C/D layout:
  // VGPR v: lanes 0-15 -> M=v, N=lane; lanes 16-31 -> M=8+v, N=lane-16
  const int mhalf = (lane < 16) ? 0 : 8;
#pragma unroll
  for (int i = 0; i < 4; ++i) {
#pragma unroll
    for (int j = 0; j < 4; ++j) {
      const int nb = n0 + wn * 64 + j * 16 + r;
      const int mb = m0 + wm * 64 + i * 16 + mhalf;
#pragma unroll
      for (int v = 0; v < 8; ++v)
        out[(size_t)(mb + v) * N_DIM + nb] = acc[i][j][v];
    }
  }
}

// =====================================================================
// Fallback: fused dequant+GEMM (round-1 kernel) if workspace too small
// =====================================================================
__global__ __launch_bounds__(256)
void awq_gemm_fused(const float* __restrict__ x,
                    const int*   __restrict__ qweight,
                    const float* __restrict__ scales,
                    const int*   __restrict__ qzeros,
                    float*       __restrict__ out) {
  __shared__ bf16_t Af[128][40];
  __shared__ bf16_t Bf[128][40];

  const int tid  = threadIdx.x;
  const int lane = tid & 31;
  const int wave = tid >> 5;
  const int wm   = wave & 3;
  const int wn   = wave >> 2;
  const int n0   = blockIdx.x * 128;
  const int m0   = blockIdx.y * 128;
  const int shifts[8] = {0, 16, 4, 20, 8, 24, 12, 28};

  v8f acc[2][4];
#pragma unroll
  for (int i = 0; i < 2; ++i)
#pragma unroll
    for (int j = 0; j < 4; ++j)
#pragma unroll
      for (int v = 0; v < 8; ++v) acc[i][j][v] = 0.0f;

  for (int k0 = 0; k0 < K_DIM; k0 += 32) {
#pragma unroll
    for (int it = 0; it < 4; ++it) {
      int u = tid + it * 256, row = u >> 3, c4 = u & 7;
      const float4* xp =
          reinterpret_cast<const float4*>(x + (size_t)(m0 + row) * K_DIM + k0) + c4;
      float4 f = *xp;
      bf16_t* dst = &Af[row][c4 * 4];
      dst[0] = f2bf(f.x); dst[1] = f2bf(f.y);
      dst[2] = f2bf(f.z); dst[3] = f2bf(f.w);
    }
    const int grp = k0 / GROUP;
#pragma unroll
    for (int it = 0; it < 2; ++it) {
      int u = tid + it * 256, row = u >> 4, cg = u & 15;
      unsigned q = (unsigned)qweight[(size_t)(k0 + row) * (N_DIM / 8) + (n0 / 8 + cg)];
      unsigned z = (unsigned)qzeros[(size_t)grp * (N_DIM / 8) + (n0 / 8 + cg)];
      const float* srow = scales + (size_t)grp * N_DIM + n0 + cg * 8;
#pragma unroll
      for (int j = 0; j < 8; ++j) {
        float wv = (float)((q >> shifts[j]) & 15u);
        float zv = (float)((z >> shifts[j]) & 15u);
        Bf[cg * 8 + j][row] = f2bf((wv - zv) * srow[j]);
      }
    }
    __syncthreads();

    v16bf afrag[2], bfrag[4];
    const int r = lane & 15;
    const int kbA = (lane < 16) ? 0 : 16;
    const int kbB = (lane < 16) ? 0 : 16;
#pragma unroll
    for (int i = 0; i < 2; ++i) {
      const bf16_t* ap = &Af[wm * 32 + i * 16 + r][0];
      B32x2 tt;
      tt.x = *reinterpret_cast<const v4u*>(ap + kbA);
      tt.y = *reinterpret_cast<const v4u*>(ap + kbA + 8);
      afrag[i] = __builtin_bit_cast(v16bf, tt);
    }
#pragma unroll
    for (int j = 0; j < 4; ++j) {
      const bf16_t* bp = &Bf[wn * 64 + j * 16 + r][kbB];
      B32x2 tt;
      tt.x = *reinterpret_cast<const v4u*>(bp);
      tt.y = *reinterpret_cast<const v4u*>(bp + 8);
      bfrag[j] = __builtin_bit_cast(v16bf, tt);
    }
#pragma unroll
    for (int i = 0; i < 2; ++i)
#pragma unroll
      for (int j = 0; j < 4; ++j)
        acc[i][j] = __builtin_amdgcn_wmma_f32_16x16x32_bf16(
            false, afrag[i], false, bfrag[j], (short)0, acc[i][j], false, false);
    __syncthreads();
  }

  const int r = lane & 15;
  const int mhalf = (lane < 16) ? 0 : 8;
#pragma unroll
  for (int i = 0; i < 2; ++i)
#pragma unroll
    for (int j = 0; j < 4; ++j) {
      const int nb = n0 + wn * 64 + j * 16 + r;
      const int mb = m0 + wm * 32 + i * 16 + mhalf;
#pragma unroll
      for (int v = 0; v < 8; ++v)
        out[(size_t)(mb + v) * N_DIM + nb] = acc[i][j][v];
    }
}

// =====================================================================
extern "C" void kernel_launch(void* const* d_in, const int* in_sizes, int n_in,
                              void* d_out, int out_size, void* d_ws, size_t ws_size,
                              hipStream_t stream) {
  const float* x       = (const float*)d_in[0];
  const int*   qweight = (const int*)d_in[1];
  const float* scales  = (const float*)d_in[2];
  const int*   qzeros  = (const int*)d_in[3];
  float* out = (float*)d_out;

  const size_t xb_bytes = (size_t)M_TOK * K_DIM * 2;   // 32 MB
  const size_t wt_bytes = (size_t)N_DIM * K_DIM * 2;   // 90 MB

  if (ws_size >= xb_bytes + wt_bytes) {
    bf16_t* xb = (bf16_t*)d_ws;
    bf16_t* wt = (bf16_t*)((char*)d_ws + xb_bytes);
    convert_x_bf16<<<(M_TOK * K_DIM / 8) / 256, 256, 0, stream>>>(x, xb);
    dequant_wt_bf16<<<dim3(K_DIM / 32, N_DIM / 64), 256, 0, stream>>>(
        qweight, scales, qzeros, wt);
    gemm_bf16_async<<<dim3(N_DIM / BN, M_TOK / BM), 256, 0, stream>>>(xb, wt, out);
  } else {
    awq_gemm_fused<<<dim3(N_DIM / 128, M_TOK / 128), 256, 0, stream>>>(
        x, qweight, scales, qzeros, out);
  }
}
